// LSTM_75419625718062
// MI455X (gfx1250) — matrix operand order repaired and checked by reading, not compiled, not run on hardware
//
#include <hip/hip_runtime.h>
#include <hip/hip_bf16.h>
#include <math.h>

// ---------------------------------------------------------------------------
// Problem constants
// ---------------------------------------------------------------------------
#define T_SEQ 512
#define BATCH 256
#define FEAT  128
#define H1    200
#define G4H   800      // 4*H1
#define HP    208      // H1 padded to 13*16 unit tiles
#define KP    224      // K (hidden) padded to 7*32 for WMMA k-chunks
#define NBLK  13       // workgroups in the recurrent kernel (unit partitions)

typedef __attribute__((ext_vector_type(16))) __bf16 v16bf;
typedef __attribute__((ext_vector_type(8)))  float  v8f;

__device__ __forceinline__ float sigf(float x) { return 1.0f / (1.0f + __expf(-x)); }

// 16-bit A-matrix 16x32 K index for (vgpr v, packed elem e, lane half)
__device__ __forceinline__ int a_k_idx(int v, int e, int half) {
    return (v < 4 ? v * 2 + e : 16 + (v - 4) * 2 + e) + half * 8;
}

// ---------------------------------------------------------------------------
// Kernel: pack w_hh1 into padded bf16 [4*HP][KP]; zero ping-pong h buffers and
// the device barrier counter.
// ---------------------------------------------------------------------------
__global__ void k_prep(const float* __restrict__ w_hh1,
                       __bf16* __restrict__ Wpk,
                       __bf16* __restrict__ hb0,
                       __bf16* __restrict__ hb1,
                       unsigned int* __restrict__ cnt) {
    int idx = blockIdx.x * 256 + threadIdx.x;
    if (idx < 4 * HP * KP) {
        int row = idx / KP, k = idx % KP;   // row = q*HP + j
        int q = row / HP, j = row % HP;
        float v = (j < H1 && k < H1) ? w_hh1[(size_t)(q * H1 + j) * H1 + k] : 0.0f;
        Wpk[idx] = (__bf16)v;
    }
    if (idx < BATCH * KP) {
        hb0[idx] = (__bf16)0.0f;
        hb1[idx] = (__bf16)0.0f;
    }
    if (idx == 0) *cnt = 0u;
}

// ---------------------------------------------------------------------------
// Kernel: xg1 = X @ w_ih1^T + (b_ih1 + b_hh1), stored bf16, shape [T*B][800].
// Block tile: 128 rows x 64 cols, 8 waves, wave w -> rowtile w x 4 coltiles.
// ---------------------------------------------------------------------------
__global__ void __launch_bounds__(256) k_xg1(const float* __restrict__ X,
                                             const float* __restrict__ w_ih1,
                                             const float* __restrict__ b_ih1,
                                             const float* __restrict__ b_hh1,
                                             __bf16* __restrict__ xg1) {
    __shared__ __bf16 As[128 * 128];
    __shared__ __bf16 Bs[64 * 128];
    __shared__ float biasS[64];

    const int tid   = threadIdx.x;
    const int mBase = blockIdx.y * 128;
    const int nBase = blockIdx.x * 64;

    // Stage A: 128 rows x 128 cols fp32 -> bf16 (4096 float4 loads)
    #pragma unroll
    for (int i = 0; i < 16; ++i) {
        int idx = tid + i * 256;
        int r = idx >> 5, c4 = idx & 31;
        const float4 f = ((const float4*)(X + (size_t)(mBase + r) * FEAT))[c4];
        As[r * 128 + c4 * 4 + 0] = (__bf16)f.x;
        As[r * 128 + c4 * 4 + 1] = (__bf16)f.y;
        As[r * 128 + c4 * 4 + 2] = (__bf16)f.z;
        As[r * 128 + c4 * 4 + 3] = (__bf16)f.w;
    }
    // Stage B: 64 gate-cols x 128 K, guard n >= 800
    #pragma unroll
    for (int i = 0; i < 8; ++i) {
        int idx = tid + i * 256;
        int n = idx >> 5, c4 = idx & 31;
        int ng = nBase + n;
        float4 f = make_float4(0.f, 0.f, 0.f, 0.f);
        if (ng < G4H) f = ((const float4*)(w_ih1 + (size_t)ng * FEAT))[c4];
        Bs[n * 128 + c4 * 4 + 0] = (__bf16)f.x;
        Bs[n * 128 + c4 * 4 + 1] = (__bf16)f.y;
        Bs[n * 128 + c4 * 4 + 2] = (__bf16)f.z;
        Bs[n * 128 + c4 * 4 + 3] = (__bf16)f.w;
    }
    if (tid < 64) {
        int ng = nBase + tid;
        biasS[tid] = (ng < G4H) ? (b_ih1[ng] + b_hh1[ng]) : 0.0f;
    }
    __syncthreads();

    const int wave = tid >> 5, lane = tid & 31;
    const int half = lane >> 4, nlo = lane & 15;

    v8f acc[4];
    #pragma unroll
    for (int ct = 0; ct < 4; ++ct)
        #pragma unroll
        for (int v = 0; v < 8; ++v) acc[ct][v] = 0.0f;

    #pragma unroll
    for (int kc = 0; kc < 4; ++kc) {
        v16bf a;
        #pragma unroll
        for (int v = 0; v < 8; ++v)
            #pragma unroll
            for (int e = 0; e < 2; ++e)
                a[v * 2 + e] = As[(wave * 16 + nlo) * 128 + kc * 32 + a_k_idx(v, e, half)];
        #pragma unroll
        for (int ct = 0; ct < 4; ++ct) {
            v16bf b;
            #pragma unroll
            for (int v = 0; v < 8; ++v)
                #pragma unroll
                for (int e = 0; e < 2; ++e)
                    b[v * 2 + e] = Bs[(ct * 16 + nlo) * 128 + kc * 32 + a_k_idx(v, e, half)];
            acc[ct] = __builtin_amdgcn_wmma_f32_16x16x32_bf16(
                false, a, false, b, (short)0, acc[ct], false, false);
        }
    }

    #pragma unroll
    for (int ct = 0; ct < 4; ++ct)
        #pragma unroll
        for (int v = 0; v < 8; ++v) {
            int m = mBase + wave * 16 + v + half * 8;
            int n = nBase + ct * 16 + nlo;
            if (n < G4H)
                xg1[(size_t)m * G4H + n] = (__bf16)(acc[ct][v] + biasS[ct * 16 + nlo]);
        }
}

// ---------------------------------------------------------------------------
// Kernel: recurrent layer-1 scan. 13 persistent blocks (one per 16-unit
// slice); W_hh slice lives in LDS for the whole kernel; c in registers.
// Double-buffered global h (bf16) + atomic-counter device barrier per step.
// h_prev staging uses CDNA5 async global->LDS DMA (ASYNCcnt).
// ---------------------------------------------------------------------------
__global__ void __launch_bounds__(256) k_lstm1(const __bf16* __restrict__ xg1,
                                               const __bf16* __restrict__ Wpk,
                                               __bf16* __restrict__ hb0,
                                               __bf16* __restrict__ hb1,
                                               float* __restrict__ h1all,
                                               unsigned int* __restrict__ cnt) {
    extern __shared__ char smem[];
    __bf16* Wl = (__bf16*)smem;                       // [4*16][KP]
    __bf16* hA = (__bf16*)(smem + 64 * KP * 2);       // [256][KP]

    const int tid = threadIdx.x;
    const int j0  = blockIdx.x * 16;                  // owned hidden units j0..j0+15

    // Load this block's W_hh slice (all 4 gates, 16 units, K padded) into LDS.
    for (int idx = tid; idx < 64 * KP; idx += 256) {
        int row = idx / KP, k = idx % KP;             // row = q*16 + jj
        int q = row >> 4, jj = row & 15;
        Wl[idx] = Wpk[((size_t)(q * HP + j0 + jj)) * KP + k];
    }

    const int wave = tid >> 5, lane = tid & 31;
    const int half = lane >> 4, nlo = lane & 15;
    const int j = j0 + nlo;

    // Wave-relative LDS byte offset of hA (generic LDS pointer: low 32 bits).
    const unsigned hA_lds = (unsigned)(uintptr_t)hA;

    float creg[2][8];
    #pragma unroll
    for (int r = 0; r < 2; ++r)
        #pragma unroll
        for (int v = 0; v < 8; ++v) creg[r][v] = 0.0f;

    for (int t = 0; t < T_SEQ; ++t) {
        const __bf16* hrd = (t & 1) ? hb1 : hb0;
        __bf16*       hwr = (t & 1) ? hb0 : hb1;

        // Stage full previous h (256 x KP bf16 = 112 KB) into LDS via
        // async global->LDS DMA; each lane moves 16B per issue.
        {
            const char* gbase = (const char*)hrd;
            for (int i = tid; i < (BATCH * KP * 2) / 16; i += 256) {
                unsigned    laddr = hA_lds + (unsigned)i * 16u;
                const void* gaddr = gbase + (size_t)i * 16;
                asm volatile("global_load_async_to_lds_b128 %0, %1, off"
                             :: "v"(laddr), "v"(gaddr)
                             : "memory");
            }
            asm volatile("s_wait_asynccnt 0x0" ::: "memory");
        }

        // Warm L2->WGP path for this/next step's gate pre-activations.
        if (t + 1 < T_SEQ && j < H1) {
            const __bf16* p = xg1 + ((size_t)(t + 1) * BATCH + (tid & 255)) * (size_t)G4H;
            __builtin_prefetch(p, 0, 1);
        }
        __syncthreads();

        v8f acc[2][4];
        #pragma unroll
        for (int r = 0; r < 2; ++r)
            #pragma unroll
            for (int q = 0; q < 4; ++q)
                #pragma unroll
                for (int v = 0; v < 8; ++v) acc[r][q][v] = 0.0f;

        #pragma unroll
        for (int rti = 0; rti < 2; ++rti) {
            const int rt = wave * 2 + rti;
            #pragma unroll
            for (int kc = 0; kc < 7; ++kc) {
                v16bf a;
                #pragma unroll
                for (int v = 0; v < 8; ++v)
                    #pragma unroll
                    for (int e = 0; e < 2; ++e)
                        a[v * 2 + e] = hA[(rt * 16 + nlo) * KP + kc * 32 + a_k_idx(v, e, half)];
                #pragma unroll
                for (int q = 0; q < 4; ++q) {
                    v16bf b;
                    #pragma unroll
                    for (int v = 0; v < 8; ++v)
                        #pragma unroll
                        for (int e = 0; e < 2; ++e)
                            b[v * 2 + e] = Wl[(q * 16 + nlo) * KP + kc * 32 + a_k_idx(v, e, half)];
                    acc[rti][q] = __builtin_amdgcn_wmma_f32_16x16x32_bf16(
                        false, a, false, b, (short)0, acc[rti][q], false, false);
                }
            }
        }

        // Gates + state update + h write-out.
        #pragma unroll
        for (int rti = 0; rti < 2; ++rti) {
            const int rt = wave * 2 + rti;
            #pragma unroll
            for (int v = 0; v < 8; ++v) {
                const int brow = rt * 16 + v + half * 8;
                float xi = 0.f, xf = 0.f, xc = 0.f, xo = 0.f;
                if (j < H1) {
                    size_t base = ((size_t)t * BATCH + brow) * (size_t)G4H;
                    xi = (float)xg1[base + j];
                    xf = (float)xg1[base + 200 + j];
                    xc = (float)xg1[base + 400 + j];
                    xo = (float)xg1[base + 600 + j];
                }
                float gi = sigf(acc[rti][0][v] + xi);
                float gf = sigf(acc[rti][1][v] + xf);
                float gc = tanhf(acc[rti][2][v] + xc);
                float go = sigf(acc[rti][3][v] + xo);
                float c  = gf * creg[rti][v] + gi * gc;
                creg[rti][v] = c;
                float h = go * tanhf(c);
                hwr[(size_t)brow * KP + j] = (__bf16)h;
                if (j < H1)
                    h1all[((size_t)t * BATCH + brow) * H1 + j] = h;
            }
        }

        // Device-wide barrier across the 13 blocks.
        __threadfence();
        __syncthreads();
        if (tid == 0) {
            const unsigned tgt = (unsigned)(NBLK * (t + 1));
            __hip_atomic_fetch_add(cnt, 1u, __ATOMIC_ACQ_REL, __HIP_MEMORY_SCOPE_AGENT);
            while (__hip_atomic_load(cnt, __ATOMIC_ACQUIRE, __HIP_MEMORY_SCOPE_AGENT) < tgt)
                __builtin_amdgcn_s_sleep(1);
        }
        __syncthreads();
    }
}

// ---------------------------------------------------------------------------
// Kernel: xg2 = h1 @ w_ih2^T + (b_ih2 + b_hh2)   (N=12, K=200; scalar fp32)
// ---------------------------------------------------------------------------
__global__ void k_xg2(const float* __restrict__ h1all,
                      const float* __restrict__ w_ih2,
                      const float* __restrict__ b_ih2,
                      const float* __restrict__ b_hh2,
                      float* __restrict__ xg2) {
    size_t idx = (size_t)blockIdx.x * 256 + threadIdx.x;
    if (idx >= (size_t)T_SEQ * BATCH * 12) return;
    int n = (int)(idx % 12);
    size_t m = idx / 12;
    float s = b_ih2[n] + b_hh2[n];
    const float* hr = h1all + m * H1;
    const float* wr = w_ih2 + (size_t)n * H1;
    #pragma unroll 4
    for (int k = 0; k < H1; ++k) s += hr[k] * wr[k];
    xg2[idx] = s;
}

// ---------------------------------------------------------------------------
// Kernel: layer-2 recurrent scan (H2=3). One block, thread b = batch lane.
// Output: (T, B*3) fp32.
// ---------------------------------------------------------------------------
__global__ void __launch_bounds__(256) k_lstm2(const float* __restrict__ xg2,
                                               const float* __restrict__ w_hh2,
                                               float* __restrict__ out) {
    const int b = threadIdx.x;
    float whh[12][3];
    #pragma unroll
    for (int n = 0; n < 12; ++n)
        #pragma unroll
        for (int m = 0; m < 3; ++m) whh[n][m] = w_hh2[n * 3 + m];

    float h[3] = {0.f, 0.f, 0.f}, c[3] = {0.f, 0.f, 0.f};
    for (int t = 0; t < T_SEQ; ++t) {
        const float* xr = xg2 + ((size_t)t * BATCH + b) * 12;
        float g[12];
        #pragma unroll
        for (int n = 0; n < 12; ++n)
            g[n] = xr[n] + h[0] * whh[n][0] + h[1] * whh[n][1] + h[2] * whh[n][2];
        #pragma unroll
        for (int u = 0; u < 3; ++u) {
            float gi = sigf(g[u]);
            float gf = sigf(g[3 + u]);
            float gc = tanhf(g[6 + u]);
            float go = sigf(g[9 + u]);
            c[u] = gf * c[u] + gi * gc;
            h[u] = go * tanhf(c[u]);
            out[(size_t)t * (BATCH * 3) + b * 3 + u] = h[u];
        }
    }
}

// ---------------------------------------------------------------------------
// Host launch
// ---------------------------------------------------------------------------
extern "C" void kernel_launch(void* const* d_in, const int* in_sizes, int n_in,
                              void* d_out, int out_size, void* d_ws, size_t ws_size,
                              hipStream_t stream) {
    (void)in_sizes; (void)n_in; (void)out_size; (void)ws_size;

    const float* X     = (const float*)d_in[0];
    const float* w_ih1 = (const float*)d_in[1];
    const float* w_hh1 = (const float*)d_in[2];
    const float* b_ih1 = (const float*)d_in[3];
    const float* b_hh1 = (const float*)d_in[4];
    const float* w_ih2 = (const float*)d_in[5];
    const float* w_hh2 = (const float*)d_in[6];
    const float* b_ih2 = (const float*)d_in[7];
    const float* b_hh2 = (const float*)d_in[8];
    float* out = (float*)d_out;

    // Workspace carving (256-byte aligned regions).
    char* ws = (char*)d_ws;
    size_t off = 0;
    auto carve = [&](size_t bytes) -> char* {
        char* p = ws + off;
        off = (off + bytes + 255) & ~(size_t)255;
        return p;
    };
    __bf16* xg1   = (__bf16*)carve((size_t)T_SEQ * BATCH * G4H * 2);   // 209.7 MB
    float*  h1all = (float*) carve((size_t)T_SEQ * BATCH * H1 * 4);    // 104.9 MB
    __bf16* Wpk   = (__bf16*)carve((size_t)4 * HP * KP * 2);           // 372 KB
    __bf16* hb0   = (__bf16*)carve((size_t)BATCH * KP * 2);            // 112 KB
    __bf16* hb1   = (__bf16*)carve((size_t)BATCH * KP * 2);            // 112 KB
    float*  xg2   = (float*) carve((size_t)T_SEQ * BATCH * 12 * 4);    // 6.3 MB
    unsigned int* cnt = (unsigned int*)carve(256);

    // 1) Pack recurrent weights + zero h ping-pong + barrier counter.
    k_prep<<<(4 * HP * KP + 255) / 256, 256, 0, stream>>>(w_hh1, Wpk, hb0, hb1, cnt);

    // 2) Big input GEMM via WMMA bf16.
    k_xg1<<<dim3((G4H + 63) / 64, (T_SEQ * BATCH) / 128), 256, 0, stream>>>(
        X, w_ih1, b_ih1, b_hh1, xg1);

    // 3) Layer-1 recurrent scan (13 persistent blocks, WMMA per step).
    const size_t smem = (size_t)(64 * KP + BATCH * KP) * 2;            // 143360 B
    k_lstm1<<<NBLK, 256, smem, stream>>>(xg1, Wpk, hb0, hb1, h1all, cnt);

    // 4) Layer-2 input contribution.
    k_xg2<<<((T_SEQ * BATCH * 12) + 255) / 256, 256, 0, stream>>>(
        h1all, w_ih2, b_ih2, b_hh2, xg2);

    // 5) Layer-2 recurrent scan.
    k_lstm2<<<1, 256, 0, stream>>>(xg2, w_hh2, out);
}
